// AveragedMedicalCLIPLoss_87290915324031
// MI455X (gfx1250) — compile-verified, matrix operand order checked by reading, not run on hardware
//
#include <hip/hip_runtime.h>
#include <hip/hip_bf16.h>

typedef __attribute__((ext_vector_type(16))) _Float16 v16h;
typedef __attribute__((ext_vector_type(8)))  _Float16 v8h;
typedef __attribute__((ext_vector_type(8)))  float    v8f;

#ifndef BDIM
#define BDIM 4096
#endif
#define DDIM 768

// ---------------------------------------------------------------------------
// K1: L2-normalize each text-embedding row, emit f16 row-major [B, D]
// ---------------------------------------------------------------------------
__global__ __launch_bounds__(256) void normalize_f16_kernel(
    const float* __restrict__ x, _Float16* __restrict__ tn, int D)
{
    const int r = blockIdx.x;
    const int t = threadIdx.x;
    const float* row = x + (size_t)r * D;

    float ss = 0.f;
    for (int d = t; d < D; d += 256) { float v = row[d]; ss += v * v; }

    __shared__ float red[256];
    red[t] = ss; __syncthreads();
    for (int s = 128; s > 0; s >>= 1) {
        if (t < s) red[t] += red[t + s];
        __syncthreads();
    }
    const float inv = rsqrtf(red[0]);

    _Float16* out = tn + (size_t)r * D;
    for (int d = t; d < D; d += 256) out[d] = (_Float16)(row[d] * inv);
}

// ---------------------------------------------------------------------------
// K2: sim = tn . tn^T via v_wmma_f32_16x16x32_f16; epilogue emits the byte
// mask (sim >= threshold). 8 waves/block, each wave: one A frag, two 16x16
// C tiles (16x32 strip). Block covers 64x64 of the 4096x4096 output.
// ---------------------------------------------------------------------------
__global__ __launch_bounds__(256) void sim_mask_kernel(
    const _Float16* __restrict__ tn, unsigned char* __restrict__ mask,
    int B, int D, float thr)
{
    const int lane  = threadIdx.x & 31;
    const int wave  = threadIdx.x >> 5;   // 0..7
    const int wm    = wave & 3;           // 4 waves along M
    const int wn    = wave >> 2;          // 2 waves along N
    const int i0    = blockIdx.y * 64 + wm * 16;
    const int j0    = blockIdx.x * 64 + wn * 32;
    const int lm    = lane & 15;
    const int khalf = (lane >> 4) * 8;    // lanes 0-15: K {0..7,16..23}; 16-31: {8..15,24..31}

    const _Float16* arow  = tn + (size_t)(i0 + lm)      * D + khalf;
    const _Float16* brow0 = tn + (size_t)(j0 + lm)      * D + khalf;
    const _Float16* brow1 = tn + (size_t)(j0 + 16 + lm) * D + khalf;

    v8f c0 = {}; v8f c1 = {};
    for (int k = 0; k < D; k += 32) {
        __builtin_prefetch(arow + k + 128, 0, 1);      // global_prefetch_b8
        v8h alo  = *(const v8h*)(arow  + k);
        v8h ahi  = *(const v8h*)(arow  + k + 16);
        v8h b0lo = *(const v8h*)(brow0 + k);
        v8h b0hi = *(const v8h*)(brow0 + k + 16);
        v8h b1lo = *(const v8h*)(brow1 + k);
        v8h b1hi = *(const v8h*)(brow1 + k + 16);
        v16h a  = __builtin_shufflevector(alo,  ahi,  0,1,2,3,4,5,6,7,8,9,10,11,12,13,14,15);
        v16h b0 = __builtin_shufflevector(b0lo, b0hi, 0,1,2,3,4,5,6,7,8,9,10,11,12,13,14,15);
        v16h b1 = __builtin_shufflevector(b1lo, b1hi, 0,1,2,3,4,5,6,7,8,9,10,11,12,13,14,15);
        c0 = __builtin_amdgcn_wmma_f32_16x16x32_f16(false, a, false, b0, (short)0, c0, false, false);
        c1 = __builtin_amdgcn_wmma_f32_16x16x32_f16(false, a, false, b1, (short)0, c1, false, false);
    }

    // C layout: VGPR k -> row i0 + k + 8*(lane>=16), col j0 + (lane&15)
    const int rbase = i0 + (lane >> 4) * 8;
    for (int k = 0; k < 8; ++k) {
        mask[(size_t)(rbase + k) * B + j0 + lm]      = (c0[k] >= thr) ? 1 : 0;
        mask[(size_t)(rbase + k) * B + j0 + 16 + lm] = (c1[k] >= thr) ? 1 : 0;
    }
}

// ---------------------------------------------------------------------------
// K3: serial greedy first-root labeling; labels + counts live in LDS (32KB).
// Single workgroup of 1024 threads; 2 barriers per row (sequential scan is
// inherent to the reference algorithm).
// ---------------------------------------------------------------------------
__global__ __launch_bounds__(1024) void label_kernel(
    const unsigned char* __restrict__ mask,
    int* __restrict__ labels_g, int* __restrict__ counts_g,
    int* __restrict__ labels_out, int B)
{
    __shared__ int lab[BDIM];
    __shared__ int cnt[BDIM];
    const int t = threadIdx.x;

    for (int j = t; j < B; j += 1024) { lab[j] = -1; cnt[j] = 0; }
    __syncthreads();

    int cur = 0;
    for (int i = 0; i < B; ++i) {
        const bool root = (lab[i] < 0);   // uniform read, pre-update
        __syncthreads();
        if (root) {
            for (int j = t; j < B; j += 1024) {
                if (j >= i && lab[j] < 0 &&
                    (j == i || mask[(size_t)i * B + j] != 0))
                    lab[j] = cur;
            }
            ++cur;
        }
        __syncthreads();
    }

    for (int j = t; j < B; j += 1024) atomicAdd(&cnt[lab[j]], 1);
    __syncthreads();
    for (int j = t; j < B; j += 1024) {
        labels_g[j]   = lab[j];
        counts_g[j]   = cnt[j];
        labels_out[j] = lab[j];
    }
}

// ---------------------------------------------------------------------------
// K4: per-row segment-mean + logsumexp over non-empty segments of
// logits_per_image (empty segments = NEG_INF contribute exactly 0 to sumexp).
// part_i[r] = lse - avg[r, labels[r]]  ( = -log_softmax target )
// ---------------------------------------------------------------------------
__global__ __launch_bounds__(256) void loss_i_kernel(
    const float* __restrict__ logits, const int* __restrict__ labels,
    const int* __restrict__ counts, float* __restrict__ part, int B)
{
    __shared__ float seg[BDIM];
    __shared__ float red[256];
    const int r = blockIdx.x;
    const int t = threadIdx.x;

    for (int s = t; s < B; s += 256) seg[s] = 0.f;
    __syncthreads();

    const float* lrow = logits + (size_t)r * B;
    for (int j = t; j < B; j += 256) atomicAdd(&seg[labels[j]], lrow[j]);
    __syncthreads();

    float mx = -3.0e38f;
    for (int s = t; s < B; s += 256)
        if (counts[s] > 0) mx = fmaxf(mx, seg[s] / (float)counts[s]);
    red[t] = mx; __syncthreads();
    for (int s2 = 128; s2 > 0; s2 >>= 1) {
        if (t < s2) red[t] = fmaxf(red[t], red[t + s2]);
        __syncthreads();
    }
    mx = red[0]; __syncthreads();

    float sum = 0.f;
    for (int s = t; s < B; s += 256)
        if (counts[s] > 0) sum += __expf(seg[s] / (float)counts[s] - mx);
    red[t] = sum; __syncthreads();
    for (int s2 = 128; s2 > 0; s2 >>= 1) {
        if (t < s2) red[t] += red[t + s2];
        __syncthreads();
    }

    if (t == 0) {
        const float lse = mx + __logf(red[0]);
        const int   lr  = labels[r];
        const float target = seg[lr] / (float)counts[lr];
        part[r] = lse - target;
    }
}

// ---------------------------------------------------------------------------
// K5: plain row logsumexp of logits_per_text.
// part_t[r] = lse - logits[r, labels[r]]
// ---------------------------------------------------------------------------
__global__ __launch_bounds__(256) void loss_t_kernel(
    const float* __restrict__ logits, const int* __restrict__ labels,
    float* __restrict__ part, int B)
{
    __shared__ float red[256];
    const int r = blockIdx.x;
    const int t = threadIdx.x;
    const float* lrow = logits + (size_t)r * B;

    float mx = -3.0e38f;
    for (int j = t; j < B; j += 256) mx = fmaxf(mx, lrow[j]);
    red[t] = mx; __syncthreads();
    for (int s = 128; s > 0; s >>= 1) {
        if (t < s) red[t] = fmaxf(red[t], red[t + s]);
        __syncthreads();
    }
    mx = red[0]; __syncthreads();

    float sum = 0.f;
    for (int j = t; j < B; j += 256) sum += __expf(lrow[j] - mx);
    red[t] = sum; __syncthreads();
    for (int s = 128; s > 0; s >>= 1) {
        if (t < s) red[t] += red[t + s];
        __syncthreads();
    }

    if (t == 0) part[r] = mx + __logf(red[0]) - lrow[labels[r]];
}

// ---------------------------------------------------------------------------
// K6: deterministic fixed-shape tree reduction -> final scalar loss
// ---------------------------------------------------------------------------
__global__ __launch_bounds__(1024) void final_kernel(
    const float* __restrict__ pi, const float* __restrict__ pt,
    float* __restrict__ out, int B)
{
    __shared__ float red[1024];
    const int t = threadIdx.x;
    float s = 0.f;
    for (int j = t; j < B; j += 1024) s += pi[j] + pt[j];
    red[t] = s; __syncthreads();
    for (int k = 512; k > 0; k >>= 1) {
        if (t < k) red[t] += red[t + k];
        __syncthreads();
    }
    if (t == 0) out[0] = 0.5f * red[0] / (float)B;
}

// ---------------------------------------------------------------------------
extern "C" void kernel_launch(void* const* d_in, const int* in_sizes, int n_in,
                              void* d_out, int out_size, void* d_ws, size_t ws_size,
                              hipStream_t stream)
{
    const int B = BDIM, D = DDIM;
    const float threshold = 0.65f;

    // inputs: 0=image_emb (unused), 1=text_emb, 2=logit_scale (unused),
    //         3=logits_per_image, 4=logits_per_text
    const float* text     = (const float*)d_in[1];
    const float* logits_i = (const float*)d_in[3];
    const float* logits_t = (const float*)d_in[4];

    char* ws = (char*)d_ws;
    _Float16*      tn     = (_Float16*)ws;      ws += (size_t)B * D * sizeof(_Float16); // 6.3 MB
    unsigned char* mask   = (unsigned char*)ws; ws += (size_t)B * B;                    // 16 MB
    int*           labels = (int*)ws;           ws += (size_t)B * sizeof(int);
    int*           counts = (int*)ws;           ws += (size_t)B * sizeof(int);
    float*         part_i = (float*)ws;         ws += (size_t)B * sizeof(float);
    float*         part_t = (float*)ws;         ws += (size_t)B * sizeof(float);

    float* out_loss   = (float*)d_out;       // d_out[0]
    int*   out_labels = (int*)d_out + 1;     // d_out[1..B] as int32 bits

    normalize_f16_kernel<<<B, 256, 0, stream>>>(text, tn, D);

    dim3 g(B / 64, B / 64);
    sim_mask_kernel<<<g, 256, 0, stream>>>(tn, mask, B, D, threshold);

    label_kernel<<<1, 1024, 0, stream>>>(mask, labels, counts, out_labels, B);

    loss_i_kernel<<<B, 256, 0, stream>>>(logits_i, labels, counts, part_i, B);
    loss_t_kernel<<<B, 256, 0, stream>>>(logits_t, labels, part_t, B);

    final_kernel<<<1, 1024, 0, stream>>>(part_i, part_t, out_loss, B);
}
